// LATTE_22247930593964
// MI455X (gfx1250) — compile-verified, matrix operand order checked by reading, not run on hardware
//
#include <hip/hip_runtime.h>
#include <hip/hip_bf16.h>

// ---------------------------------------------------------------------------
// LATTE forward for MI455X (gfx1250). Structure:
//   K1: l = x@Wl^T+bl, r = x@Wr^T+br via split-bf16 WMMA (fp32-accurate-ish)
//   K2: per-node per-head attention scores (pre-multiplied by sharp[m])
//   K3: beta = softmax(x@conv_w^T + conv_b)
//   K4: out = beta[:,3] * r            (self relation)
//   per metapath m: K5 init, K6 atomic-max, K7 atomic-denom,
//                   K8 alpha-weighted scatter-add, K9 out += beta[:,m]*outm
//   (ReLU folded into last K9)
// ---------------------------------------------------------------------------

#define HEADS 4
#define CH    32
#define DDIM  128
#define NSLOPE 0.2f

typedef __attribute__((ext_vector_type(16))) __bf16 v16bf;
typedef __attribute__((ext_vector_type(8)))  float  v8f;

// ---- ordered-uint encoding for atomic float max -----------------------------
__device__ __forceinline__ unsigned f32_enc(float f) {
  unsigned b = __float_as_uint(f);
  return (b & 0x80000000u) ? ~b : (b | 0x80000000u);
}
__device__ __forceinline__ float f32_dec(unsigned u) {
  unsigned b = (u & 0x80000000u) ? (u ^ 0x80000000u) : ~u;
  return __uint_as_float(b);
}
#define ENC_NEG_INF 0x007FFFFFu   // f32_enc(-inf)

__device__ __forceinline__ void atomAddF(float* p, float v) {
  unsafeAtomicAdd(p, v);          // lowers to global_atomic_add_f32
}

// ---- split a fp32 stream into hi/lo bf16 WMMA A/B fragments -----------------
// p points at (row base + kb + k0); element i -> K = k0+kb+i, element 8+i -> K+16
__device__ __forceinline__ void load_frag_split(const float* __restrict__ p,
                                                v16bf& hi, v16bf& lo) {
#pragma unroll
  for (int i = 0; i < 8; ++i) {
    float f0 = p[i];
    float f1 = p[16 + i];
    __bf16 h0 = (__bf16)f0;
    __bf16 h1 = (__bf16)f1;
    hi[i]     = h0;
    hi[8 + i] = h1;
    lo[i]     = (__bf16)(f0 - (float)h0);
    lo[8 + i] = (__bf16)(f1 - (float)h1);
  }
}

__device__ __forceinline__ v8f wmma_bf16(v16bf a, v16bf b, v8f c) {
  return __builtin_amdgcn_wmma_f32_16x16x32_bf16(
      /*neg_a=*/false, a, /*neg_b=*/false, b,
      /*c_mod=*/(short)0, c, /*reuse_a=*/false, /*reuse_b=*/false);
}

// ---------------------------------------------------------------------------
// K1: dual GEMM, one wave = one 16x16 tile of both l and r.
// grid.x = N/16 row tiles; 8 waves/block cover the 8 column tiles.
// ---------------------------------------------------------------------------
__global__ void gemm_lr_kernel(const float* __restrict__ x,
                               const float* __restrict__ Wl,
                               const float* __restrict__ bl,
                               const float* __restrict__ Wr,
                               const float* __restrict__ br,
                               float* __restrict__ l_out,
                               float* __restrict__ r_out) {
  const int wave = threadIdx.x >> 5;        // 0..7 column tile
  const int lane = threadIdx.x & 31;
  const int row0 = blockIdx.x * 16;
  const int col0 = wave * 16;
  const int m    = lane & 15;
  const int kb   = (lane >> 4) * 8;         // K sub-offset per half-wave

  const float* xrow  = x  + (size_t)(row0 + m) * DDIM + kb;   // A: row = m
  const float* wlrow = Wl + (size_t)(col0 + m) * DDIM + kb;   // B: lane idx = out chan
  const float* wrrow = Wr + (size_t)(col0 + m) * DDIM + kb;

  v8f cl = {}; v8f cr = {};
#pragma unroll
  for (int k0 = 0; k0 < DDIM; k0 += 32) {
    v16bf a_hi, a_lo, pbh, pbl, qbh, qbl;
    load_frag_split(xrow  + k0, a_hi, a_lo);
    load_frag_split(wlrow + k0, pbh,  pbl);
    load_frag_split(wrrow + k0, qbh,  qbl);
    // x@W ~= xh@Wh + xh@Wl + xl@Wh  (split-bf16, fp32 accumulate)
    cl = wmma_bf16(a_hi, pbh, cl);
    cl = wmma_bf16(a_hi, pbl, cl);
    cl = wmma_bf16(a_lo, pbh, cl);
    cr = wmma_bf16(a_hi, qbh, cr);
    cr = wmma_bf16(a_hi, qbl, cr);
    cr = wmma_bf16(a_lo, qbh, cr);
  }

  // C/D layout: VGPR v -> M = v (lanes 0-15) / 8+v (lanes 16-31), N = lane&15
  const int col   = col0 + m;
  const float blv = bl[col];
  const float brv = br[col];
  const int rbase = row0 + (lane >> 4) * 8;
#pragma unroll
  for (int v = 0; v < 8; ++v) {
    l_out[(size_t)(rbase + v) * DDIM + col] = cl[v] + blv;
    r_out[(size_t)(rbase + v) * DDIM + col] = cr[v] + brv;
  }
}

// ---------------------------------------------------------------------------
// K2: per-node/head scores, pre-scaled by sharp[m].  a_* layout: [M][N][H]
// ---------------------------------------------------------------------------
__global__ void scores_kernel(const float* __restrict__ l,
                              const float* __restrict__ r,
                              const float* __restrict__ attn,   // [3][4][64]
                              const float* __restrict__ sharp,  // [3]
                              float* __restrict__ a_src,
                              float* __restrict__ a_dst,
                              int n_nodes) {
  const int t = blockIdx.x * blockDim.x + threadIdx.x;
  if (t >= n_nodes * HEADS) return;
  const int n = t >> 2, h = t & 3;
  const float* lp = l + (size_t)n * DDIM + h * CH;
  const float* rp = r + (size_t)n * DDIM + h * CH;
  float accs[3] = {0.f, 0.f, 0.f};
  float accd[3] = {0.f, 0.f, 0.f};
#pragma unroll 4
  for (int c = 0; c < CH; ++c) {
    float lv = lp[c]; lv = lv > 0.f ? lv : NSLOPE * lv;   // leaky_relu(l)
    float rv = rp[c]; rv = rv > 0.f ? rv : NSLOPE * rv;   // leaky_relu(r)
#pragma unroll
    for (int mm = 0; mm < 3; ++mm) {
      const float* ap = attn + (size_t)(mm * HEADS + h) * (2 * CH);
      accd[mm] = fmaf(rv, ap[c],      accd[mm]);   // attn[m,h,:C]   (dst)
      accs[mm] = fmaf(lv, ap[CH + c], accs[mm]);   // attn[m,h,C:]   (src)
    }
  }
#pragma unroll
  for (int mm = 0; mm < 3; ++mm) {
    const float s = sharp[mm];
    a_src[(size_t)mm * n_nodes * HEADS + t] = s * accs[mm];
    a_dst[(size_t)mm * n_nodes * HEADS + t] = s * accd[mm];
  }
}

// ---------------------------------------------------------------------------
// K3: beta = softmax(x @ conv_w^T + conv_b) over 4 relations
// ---------------------------------------------------------------------------
__global__ void beta_kernel(const float* __restrict__ x,
                            const float* __restrict__ cw,   // [4][128]
                            const float* __restrict__ cb,   // [4]
                            float* __restrict__ beta,
                            int n_nodes) {
  const int n = blockIdx.x * blockDim.x + threadIdx.x;
  if (n >= n_nodes) return;
  const float* xp = x + (size_t)n * DDIM;
  float acc[4] = {cb[0], cb[1], cb[2], cb[3]};
  for (int k = 0; k < DDIM; k += 4) {
    const float4 xv = *(const float4*)(xp + k);
#pragma unroll
    for (int rr = 0; rr < 4; ++rr) {
      const float* w = cw + rr * DDIM + k;
      acc[rr] = fmaf(xv.x, w[0], fmaf(xv.y, w[1], fmaf(xv.z, w[2], fmaf(xv.w, w[3], acc[rr]))));
    }
  }
  float mx = fmaxf(fmaxf(acc[0], acc[1]), fmaxf(acc[2], acc[3]));
  float sum = 0.f;
#pragma unroll
  for (int rr = 0; rr < 4; ++rr) { acc[rr] = __expf(acc[rr] - mx); sum += acc[rr]; }
  const float inv = 1.f / sum;
#pragma unroll
  for (int rr = 0; rr < 4; ++rr) beta[(size_t)n * 4 + rr] = acc[rr] * inv;
}

// K4: out = beta[:,3] * r   (also clears d_out poison)
__global__ void init_out_kernel(const float* __restrict__ r,
                                const float* __restrict__ beta,
                                float* __restrict__ out, int n_nodes) {
  const int t = blockIdx.x * blockDim.x + threadIdx.x;   // N*32 float4 groups
  if (t >= n_nodes * (DDIM / 4)) return;
  const int n = t >> 5;
  const float b = beta[(size_t)n * 4 + 3];
  float4 v = *(const float4*)(r + (size_t)t * 4);
  v.x *= b; v.y *= b; v.z *= b; v.w *= b;
  *(float4*)(out + (size_t)t * 4) = v;
}

// K5: zero per-metapath accumulators
__global__ void init_node_kernel(unsigned* __restrict__ nmax,
                                 float* __restrict__ denom,
                                 float* __restrict__ outm, int n_nodes) {
  const int t = blockIdx.x * blockDim.x + threadIdx.x;
  if (t < n_nodes * HEADS) { nmax[t] = ENC_NEG_INF; denom[t] = 0.f; }
  if (t < n_nodes * DDIM)  outm[t] = 0.f;
}

// K6: scatter-max of edge scores onto destination nodes
__global__ void edge_max_kernel(const int* __restrict__ src,
                                const int* __restrict__ dst,
                                const float* __restrict__ as,
                                const float* __restrict__ ad,
                                unsigned* __restrict__ nmax, int n_edges) {
  const int t = blockIdx.x * blockDim.x + threadIdx.x;   // E*H
  if (t >= n_edges * HEADS) return;
  const int e = t >> 2, h = t & 3;
  const int s = src[e], d = dst[e];
  const float sc = ad[(size_t)d * HEADS + h] + as[(size_t)s * HEADS + h];
  atomicMax(&nmax[(size_t)d * HEADS + h], f32_enc(sc));
}

// K7: denom += exp(score - max[dst])
__global__ void edge_denom_kernel(const int* __restrict__ src,
                                  const int* __restrict__ dst,
                                  const float* __restrict__ as,
                                  const float* __restrict__ ad,
                                  const unsigned* __restrict__ nmax,
                                  float* __restrict__ denom, int n_edges) {
  const int t = blockIdx.x * blockDim.x + threadIdx.x;
  if (t >= n_edges * HEADS) return;
  const int e = t >> 2, h = t & 3;
  const int s = src[e], d = dst[e];
  const size_t d4 = (size_t)d * HEADS + h;
  const float sc = ad[d4] + as[(size_t)s * HEADS + h];
  atomAddF(&denom[d4], __expf(sc - f32_dec(nmax[d4])));
}

// K8: outm[dst] += alpha * l[src]   (32 channels per thread, fp32 atomics)
__global__ void edge_scatter_kernel(const int* __restrict__ src,
                                    const int* __restrict__ dst,
                                    const float* __restrict__ as,
                                    const float* __restrict__ ad,
                                    const unsigned* __restrict__ nmax,
                                    const float* __restrict__ denom,
                                    const float* __restrict__ l,
                                    float* __restrict__ outm, int n_edges) {
  const int t = blockIdx.x * blockDim.x + threadIdx.x;
  if (t >= n_edges * HEADS) return;
  const int e = t >> 2, h = t & 3;
  const int s = src[e], d = dst[e];
  const size_t d4 = (size_t)d * HEADS + h;
  const float sc    = ad[d4] + as[(size_t)s * HEADS + h];
  const float alpha = __expf(sc - f32_dec(nmax[d4])) / denom[d4];
  const float* lp = l    + (size_t)s * DDIM + h * CH;
  float*       op = outm + (size_t)d * DDIM + h * CH;
#pragma unroll
  for (int c = 0; c < CH; c += 4) {
    const float4 lv = *(const float4*)(lp + c);
    atomAddF(op + c + 0, lv.x * alpha);
    atomAddF(op + c + 1, lv.y * alpha);
    atomAddF(op + c + 2, lv.z * alpha);
    atomAddF(op + c + 3, lv.w * alpha);
  }
}

// K9: out += beta[:,m] * outm ; optional final ReLU
__global__ void accum_kernel(const float* __restrict__ outm,
                             const float* __restrict__ beta,
                             float* __restrict__ out,
                             int mslot, int do_relu, int n_nodes) {
  const int t = blockIdx.x * blockDim.x + threadIdx.x;   // N*32 float4 groups
  if (t >= n_nodes * (DDIM / 4)) return;
  const int n = t >> 5;
  const float b = beta[(size_t)n * 4 + mslot];
  const float4 v = *(const float4*)(outm + (size_t)t * 4);
  float4 o = *(const float4*)(out + (size_t)t * 4);
  o.x = fmaf(b, v.x, o.x); o.y = fmaf(b, v.y, o.y);
  o.z = fmaf(b, v.z, o.z); o.w = fmaf(b, v.w, o.w);
  if (do_relu) {
    o.x = fmaxf(o.x, 0.f); o.y = fmaxf(o.y, 0.f);
    o.z = fmaxf(o.z, 0.f); o.w = fmaxf(o.w, 0.f);
  }
  *(float4*)(out + (size_t)t * 4) = o;
}

// ---------------------------------------------------------------------------
static inline unsigned cdiv(long long a, long long b) { return (unsigned)((a + b - 1) / b); }

extern "C" void kernel_launch(void* const* d_in, const int* in_sizes, int n_in,
                              void* d_out, int out_size, void* d_ws, size_t ws_size,
                              hipStream_t stream) {
  const float* x      = (const float*)d_in[0];
  const float* Wl     = (const float*)d_in[1];
  const float* bl     = (const float*)d_in[2];
  const float* Wr     = (const float*)d_in[3];
  const float* br     = (const float*)d_in[4];
  const float* attn   = (const float*)d_in[5];
  const float* sharp  = (const float*)d_in[6];
  const float* conv_w = (const float*)d_in[7];
  const float* conv_b = (const float*)d_in[8];

  const int N = in_sizes[0] / DDIM;       // 100000
  const int E = in_sizes[9] / 2;          // 1600000

  // Workspace layout (floats): l, r, a_src[3], a_dst[3], beta, nmax, denom, outm
  float* ws     = (float*)d_ws;
  float* l      = ws;
  float* r      = l     + (size_t)N * DDIM;
  float* a_src  = r     + (size_t)N * DDIM;
  float* a_dst  = a_src + (size_t)3 * N * HEADS;
  float* beta   = a_dst + (size_t)3 * N * HEADS;
  unsigned* nmax = (unsigned*)(beta + (size_t)N * HEADS);
  float* denom  = (float*)(nmax + (size_t)N * HEADS);
  float* outm   = denom + (size_t)N * HEADS;
  float* out    = (float*)d_out;

  const int BLK = 256;

  // K1: dual WMMA GEMM (N/16 row tiles x 8 col-tile waves)
  gemm_lr_kernel<<<dim3(cdiv(N, 16)), dim3(256), 0, stream>>>(x, Wl, bl, Wr, br, l, r);

  // K2: attention scores (all 3 metapaths at once)
  scores_kernel<<<dim3(cdiv((long long)N * HEADS, BLK)), dim3(BLK), 0, stream>>>(
      l, r, attn, sharp, a_src, a_dst, N);

  // K3: beta softmax
  beta_kernel<<<dim3(cdiv(N, BLK)), dim3(BLK), 0, stream>>>(x, conv_w, conv_b, beta, N);

  // K4: out = beta[:,3] * r
  init_out_kernel<<<dim3(cdiv((long long)N * (DDIM / 4), BLK)), dim3(BLK), 0, stream>>>(
      r, beta, out, N);

  const unsigned eh_grid = cdiv((long long)E * HEADS, BLK);
  for (int mm = 0; mm < 3; ++mm) {
    const int* ei  = (const int*)d_in[9 + mm];
    const int* srcp = ei;
    const int* dstp = ei + E;
    const float* as = a_src + (size_t)mm * N * HEADS;
    const float* ad = a_dst + (size_t)mm * N * HEADS;

    init_node_kernel<<<dim3(cdiv((long long)N * DDIM, BLK)), dim3(BLK), 0, stream>>>(
        nmax, denom, outm, N);
    edge_max_kernel<<<dim3(eh_grid), dim3(BLK), 0, stream>>>(srcp, dstp, as, ad, nmax, E);
    edge_denom_kernel<<<dim3(eh_grid), dim3(BLK), 0, stream>>>(srcp, dstp, as, ad, nmax, denom, E);
    edge_scatter_kernel<<<dim3(eh_grid), dim3(BLK), 0, stream>>>(
        srcp, dstp, as, ad, nmax, denom, l, outm, E);
    accum_kernel<<<dim3(cdiv((long long)N * (DDIM / 4), BLK)), dim3(BLK), 0, stream>>>(
        outm, beta, out, mm, /*do_relu=*/(mm == 2) ? 1 : 0, N);
  }
}
